// CLSGuidedMoE_1262720385565
// MI455X (gfx1250) — matrix operand order repaired and checked by reading, not compiled, not run on hardware
//
#include <hip/hip_runtime.h>
#include <hip/hip_bf16.h>

typedef __bf16 bf16_t;
typedef __bf16 v16bf __attribute__((ext_vector_type(16)));
typedef __bf16 v8bf  __attribute__((ext_vector_type(8)));
typedef float  v8f   __attribute__((ext_vector_type(8)));

#define B_ROWS 8192
#define FEATD  1024
#define NHEAD  16
#define NEXP   7
#define HDIM   64
#define OUTC   (NEXP * FEATD)   // 7168
#define NSLICE 64
#define ROWS_PER_SLICE (B_ROWS / NSLICE) // 128

// ---------------------------------------------------------------- helpers

__device__ __forceinline__ float quickgelu(float x) {
    return x / (1.0f + __expf(-1.702f * x));
}

__device__ __forceinline__ v8f zero8() {
    v8f z;
#pragma unroll
    for (int i = 0; i < 8; ++i) z[i] = 0.0f;
    return z;
}

// A-matrix fragment, 16x32 bf16 (M x K), row-major source with leading dim ld.
// Lane L<16: row M=L, K = [0..7] then [16..23]; lane L>=16: row M=L-16, K=[8..15],[24..31].
__device__ __forceinline__ v16bf load_fragA(const bf16_t* __restrict__ tile, int ld, int lane) {
    const int r  = lane & 15;
    const int ko = (lane >> 4) << 3;
    const bf16_t* p = tile + (size_t)r * ld + ko;
    v8bf lo = *(const v8bf*)(p);
    v8bf hi = *(const v8bf*)(p + 16);
    v16bf f;
#pragma unroll
    for (int i = 0; i < 8; ++i) { f[i] = lo[i]; f[i + 8] = hi[i]; }
    return f;
}

// B-matrix fragment, 32x16 bf16 (K x N), sourced from weight row-major (N x K, ld=K):
// lane L<16: col N=L, K = [0..7] then [16..23]; lane L>=16: col N=L-16, K=[8..15],[24..31].
__device__ __forceinline__ v16bf load_fragB(const bf16_t* __restrict__ tile, int ld, int lane) {
    const int n  = lane & 15;
    const int ko = (lane >> 4) << 3;
    const bf16_t* p = tile + (size_t)n * ld + ko;
    v8bf lo = *(const v8bf*)(p);
    v8bf hi = *(const v8bf*)(p + 16);
    v16bf f;
#pragma unroll
    for (int i = 0; i < 8; ++i) { f[i] = lo[i]; f[i + 8] = hi[i]; }
    return f;
}

__device__ __forceinline__ v8f wmma_bf16(v16bf a, v16bf b, v8f c) {
    return __builtin_amdgcn_wmma_f32_16x16x32_bf16(false, a, false, b, (short)0, c, false, false);
}

// ---------------------------------------------------------------- converts

__global__ void cvt_bf16(const float* __restrict__ src, bf16_t* __restrict__ dst, size_t n) {
    size_t i = (size_t)blockIdx.x * blockDim.x + threadIdx.x;
    size_t stride = (size_t)gridDim.x * blockDim.x;
    for (; i < n; i += stride) dst[i] = (bf16_t)src[i];
}

// ---------------------------------------------------------------- stage A: query = quickgelu([cls|x] @ W_re^T + b_re)
// Wave tile: 64(M) x 64(N) = 4x4 fragments -> 16 WMMA per 16 b128 loads per K-step.
// __launch_bounds__(256, 1): grant full per-wave VGPR budget (no accumulator spills).

__global__ __launch_bounds__(256, 1)
void gemm_gate(const bf16_t* __restrict__ Xcat,   // 8 chunks of (8192x1024) bf16, contiguous
               const bf16_t* __restrict__ Wre,    // (1024 x 8192) bf16 row-major
               const float*  __restrict__ bias,   // (1024)
               float* __restrict__ C) {           // (8192 x 1024) f32
    const int lane = threadIdx.x & 31;
    const int wid  = (blockIdx.x * blockDim.x + threadIdx.x) >> 5;
    const int nTiles = FEATD / 64;              // 16
    const int mT = wid / nTiles;
    const int nT = wid % nTiles;
    if (mT >= B_ROWS / 64) return;
    const int row0 = mT * 64;
    const int col0 = nT * 64;

    v8f acc[4][4];
#pragma unroll
    for (int i = 0; i < 4; ++i)
#pragma unroll
        for (int j = 0; j < 4; ++j) acc[i][j] = zero8();

    for (int kb = 0; kb < 8 * FEATD; kb += 32) {
        const int chunk = kb >> 10;
        const int cc    = kb & (FEATD - 1);
        const bf16_t* Abase = Xcat + (size_t)chunk * ((size_t)B_ROWS * FEATD)
                                   + (size_t)row0 * FEATD + cc;
        v16bf a[4];
#pragma unroll
        for (int mf = 0; mf < 4; ++mf)
            a[mf] = load_fragA(Abase + (size_t)(mf * 16) * FEATD, FEATD, lane);
#pragma unroll
        for (int nf = 0; nf < 4; ++nf) {
            v16bf b = load_fragB(Wre + (size_t)(col0 + nf * 16) * (8 * FEATD) + kb, 8 * FEATD, lane);
#pragma unroll
            for (int mf = 0; mf < 4; ++mf)
                acc[mf][nf] = wmma_bf16(a[mf], b, acc[mf][nf]);
        }
    }

    const int nlane = lane & 15;
    const int rbase = (lane >> 4) << 3;
#pragma unroll
    for (int mf = 0; mf < 4; ++mf)
#pragma unroll
        for (int nf = 0; nf < 4; ++nf) {
            const int col = col0 + nf * 16 + nlane;
            const float bv = bias[col];
#pragma unroll
            for (int r = 0; r < 8; ++r) {
                const int row = row0 + mf * 16 + rbase + r;
                C[(size_t)row * FEATD + col] = quickgelu(acc[mf][nf][r] + bv);
            }
        }
}

// ---------------------------------------------------------------- plain GEMM: C = A(8192xK) @ W^T, f32 out
// grid.y selects batched operand (used for the 7 key GEMMs). 64x64 wave tile.

__global__ __launch_bounds__(256, 1)
void gemm_nt(const bf16_t* __restrict__ A0, size_t aStride,
             const bf16_t* __restrict__ W,
             float* __restrict__ C0, size_t cStride, int ldc,
             int nTiles, int K) {
    const bf16_t* A = A0 + (size_t)blockIdx.y * aStride;
    float*        C = C0 + (size_t)blockIdx.y * cStride;
    const int lane = threadIdx.x & 31;
    const int wid  = (blockIdx.x * blockDim.x + threadIdx.x) >> 5;
    const int mT = wid / nTiles;
    const int nT = wid % nTiles;
    if (mT >= B_ROWS / 64) return;
    const int row0 = mT * 64;
    const int col0 = nT * 64;

    v8f acc[4][4];
#pragma unroll
    for (int i = 0; i < 4; ++i)
#pragma unroll
        for (int j = 0; j < 4; ++j) acc[i][j] = zero8();

    for (int kb = 0; kb < K; kb += 32) {
        const bf16_t* Abase = A + (size_t)row0 * K + kb;
        v16bf a[4];
#pragma unroll
        for (int mf = 0; mf < 4; ++mf)
            a[mf] = load_fragA(Abase + (size_t)(mf * 16) * K, K, lane);
#pragma unroll
        for (int nf = 0; nf < 4; ++nf) {
            v16bf b = load_fragB(W + (size_t)(col0 + nf * 16) * K + kb, K, lane);
#pragma unroll
            for (int mf = 0; mf < 4; ++mf)
                acc[mf][nf] = wmma_bf16(a[mf], b, acc[mf][nf]);
        }
    }

    const int nlane = lane & 15;
    const int rbase = (lane >> 4) << 3;
#pragma unroll
    for (int mf = 0; mf < 4; ++mf)
#pragma unroll
        for (int nf = 0; nf < 4; ++nf) {
            const int col = col0 + nf * 16 + nlane;
#pragma unroll
            for (int r = 0; r < 8; ++r) {
                const int row = row0 + mf * 16 + rbase + r;
                C[(size_t)row * ldc + col] = acc[mf][nf][r];
            }
        }
}

// ---------------------------------------------------------------- expert GEMMs: 112 x (8192x64 @ 64x64^T), +bias, quickgelu
// K=64 (2 steps), weights are tiny and L0-resident; 32x64 wave tile suffices.

__global__ __launch_bounds__(256, 1)
void gemm_expert(const bf16_t* __restrict__ X1,    // x1..x7 bf16 chunks, contiguous
                 const bf16_t* __restrict__ WeBf,  // (112, 64, 64) bf16
                 const float*  __restrict__ be,    // (112, 64)
                 float* __restrict__ Out) {        // (8192, 7168) f32
    const int lane = threadIdx.x & 31;
    const int wid  = (blockIdx.x * blockDim.x + threadIdx.x) >> 5;
    const int mT = wid & 255;          // 256 m-tiles of 32 rows
    const int he = wid >> 8;           // h*7 + e
    if (he >= NHEAD * NEXP) return;
    const int h = he / NEXP;
    const int e = he % NEXP;
    const bf16_t* A  = X1 + (size_t)e * ((size_t)B_ROWS * FEATD) + h * HDIM;
    const bf16_t* Wp = WeBf + (size_t)he * (HDIM * HDIM);
    const int row0 = mT * 32;

    v8f acc[2][4];
#pragma unroll
    for (int i = 0; i < 2; ++i)
#pragma unroll
        for (int j = 0; j < 4; ++j) acc[i][j] = zero8();

#pragma unroll
    for (int kb = 0; kb < HDIM; kb += 32) {
        const bf16_t* Abase = A + (size_t)row0 * FEATD + kb;
        v16bf a0 = load_fragA(Abase, FEATD, lane);
        v16bf a1 = load_fragA(Abase + 16 * FEATD, FEATD, lane);
#pragma unroll
        for (int nf = 0; nf < 4; ++nf) {
            v16bf b = load_fragB(Wp + (size_t)(nf * 16) * HDIM + kb, HDIM, lane);
            acc[0][nf] = wmma_bf16(a0, b, acc[0][nf]);
            acc[1][nf] = wmma_bf16(a1, b, acc[1][nf]);
        }
    }

    const int nlane = lane & 15;
    const int rbase = (lane >> 4) << 3;
#pragma unroll
    for (int mf = 0; mf < 2; ++mf)
#pragma unroll
        for (int nf = 0; nf < 4; ++nf) {
            const int o  = nf * 16 + nlane;           // local out feature
            const float bv = be[he * HDIM + o];
            const int col = e * FEATD + h * HDIM + o; // output column
#pragma unroll
            for (int r = 0; r < 8; ++r) {
                const int row = row0 + mf * 16 + rbase + r;
                Out[(size_t)row * OUTC + col] = quickgelu(acc[mf][nf][r] + bv);
            }
        }
}

// ---------------------------------------------------------------- BatchNorm stats (deterministic two-phase)

__global__ void colstats_partial(const float* __restrict__ M, int ld, int cols,
                                 float* __restrict__ psum, float* __restrict__ psq) {
    const int idx = blockIdx.x * blockDim.x + threadIdx.x;
    if (idx >= cols * NSLICE) return;
    const int c = idx % cols;
    const int s = idx / cols;
    const float* p = M + (size_t)s * ROWS_PER_SLICE * ld + c;
    float sum = 0.0f, sq = 0.0f;
    for (int r = 0; r < ROWS_PER_SLICE; ++r) {
        const float v = p[(size_t)r * ld];
        sum += v; sq += v * v;
    }
    psum[idx] = sum;
    psq[idx]  = sq;
}

__global__ void colstats_final(const float* __restrict__ psum, const float* __restrict__ psq,
                               int cols, const float* __restrict__ gamma,
                               const float* __restrict__ beta, int mode,
                               float* __restrict__ a, float* __restrict__ cshift) {
    const int c = blockIdx.x * blockDim.x + threadIdx.x;
    if (c >= cols) return;
    float s = 0.0f, q = 0.0f;
    for (int i = 0; i < NSLICE; ++i) {
        s += psum[(size_t)i * cols + c];
        q += psq[(size_t)i * cols + c];
    }
    const float invB = 1.0f / (float)B_ROWS;
    const float mean = s * invB;
    const float var  = fmaxf(q * invB - mean * mean, 0.0f);
    int gi = c;
    if (mode == 1) { // out column e*1024+h*64+d  ->  param index (h*7+e)*64+d
        const int e = c / FEATD;
        const int rem = c - e * FEATD;
        const int h = rem >> 6;
        const int d = rem & 63;
        gi = (h * NEXP + e) * HDIM + d;
    }
    const float aa = gamma[gi] * rsqrtf(var + 1e-5f);
    a[c] = aa;
    cshift[c] = beta[gi] - mean * aa;
}

__global__ void bn_apply_bf16(const float* __restrict__ src, const float* __restrict__ a,
                              const float* __restrict__ c, bf16_t* __restrict__ dst, size_t n) {
    size_t i = (size_t)blockIdx.x * blockDim.x + threadIdx.x;
    size_t stride = (size_t)gridDim.x * blockDim.x;
    for (; i < n; i += stride) {
        const int col = (int)(i & (FEATD - 1));
        dst[i] = (bf16_t)(src[i] * a[col] + c[col]);
    }
}

// ---------------------------------------------------------------- attention logits + softmax gates

__global__ void attn_softmax(const float* __restrict__ q,   // (8192,1024)
                             const float* __restrict__ k,   // (8192,7,1024)
                             float* __restrict__ gates) {   // (8192,16,7)
    const int t = blockIdx.x * blockDim.x + threadIdx.x;
    if (t >= B_ROWS * NHEAD) return;
    const int b = t >> 4;
    const int h = t & 15;
    const float* qp = q + (size_t)b * FEATD + h * HDIM;
    float logit[NEXP];
    float mx = -1e30f;
#pragma unroll
    for (int e = 0; e < NEXP; ++e) {
        const float* kp = k + (size_t)b * OUTC + e * FEATD + h * HDIM;
        float d = 0.0f;
#pragma unroll 8
        for (int i = 0; i < HDIM; ++i) d += qp[i] * kp[i];
        d *= 0.125f; // 64^-0.5
        logit[e] = d;
        mx = fmaxf(mx, d);
    }
    float s = 0.0f;
#pragma unroll
    for (int e = 0; e < NEXP; ++e) { logit[e] = __expf(logit[e] - mx); s += logit[e]; }
    const float inv = 1.0f / s;
#pragma unroll
    for (int e = 0; e < NEXP; ++e) gates[(size_t)t * NEXP + e] = logit[e] * inv;
}

// ---------------------------------------------------------------- final BN + gate

__global__ void finalize(float* __restrict__ out, const float* __restrict__ a,
                         const float* __restrict__ c, const float* __restrict__ gates, size_t n) {
    size_t i = (size_t)blockIdx.x * blockDim.x + threadIdx.x;
    size_t stride = (size_t)gridDim.x * blockDim.x;
    for (; i < n; i += stride) {
        const int col = (int)(i % OUTC);
        const size_t b = i / OUTC;
        const int e = col / FEATD;
        const int rem = col - e * FEATD;
        const int h = rem >> 6;
        const float g = gates[(b * NHEAD + h) * NEXP + e];
        out[i] = (out[i] * a[col] + c[col]) * g;
    }
}

// ---------------------------------------------------------------- host

extern "C" void kernel_launch(void* const* d_in, const int* in_sizes, int n_in,
                              void* d_out, int out_size, void* d_ws, size_t ws_size,
                              hipStream_t stream) {
    (void)in_sizes; (void)n_in; (void)out_size; (void)ws_size;

    const float* cls  = (const float*)d_in[0];
    const float* xs[NEXP];
    for (int e = 0; e < NEXP; ++e) xs[e] = (const float*)d_in[1 + e];
    const float* W_re = (const float*)d_in[8];
    const float* b_re = (const float*)d_in[9];
    const float* g_re = (const float*)d_in[10];
    const float* bb_re= (const float*)d_in[11];
    const float* Wq   = (const float*)d_in[12];
    const float* Wk   = (const float*)d_in[13];
    const float* We   = (const float*)d_in[14];
    const float* be   = (const float*)d_in[15];
    const float* g_e  = (const float*)d_in[16];
    const float* bb_e = (const float*)d_in[17];

    char* ws = (char*)d_ws;
    size_t off = 0;
    auto alloc = [&](size_t bytes) -> char* {
        char* p = ws + off;
        off += (bytes + 255) & ~(size_t)255;
        return p;
    };

    const size_t MAT = (size_t)B_ROWS * FEATD;               // 8M elements
    bf16_t* Xbf   = (bf16_t*)alloc(8 * MAT * sizeof(bf16_t));          // [cls|x1..x7] bf16
    bf16_t* WreBf = (bf16_t*)alloc((size_t)FEATD * 8 * FEATD * sizeof(bf16_t));
    bf16_t* WqBf  = (bf16_t*)alloc((size_t)FEATD * FEATD * sizeof(bf16_t));
    bf16_t* WkBf  = (bf16_t*)alloc((size_t)FEATD * FEATD * sizeof(bf16_t));
    bf16_t* WeBf  = (bf16_t*)alloc((size_t)NHEAD * NEXP * HDIM * HDIM * sizeof(bf16_t));
    float*  query = (float*)alloc(MAT * sizeof(float));                // reused as q after BN
    bf16_t* qbnBf = (bf16_t*)alloc(MAT * sizeof(bf16_t));
    float*  kmat  = (float*)alloc((size_t)B_ROWS * OUTC * sizeof(float));
    float*  gates = (float*)alloc((size_t)B_ROWS * NHEAD * NEXP * sizeof(float));
    float*  a1    = (float*)alloc(FEATD * sizeof(float));
    float*  c1    = (float*)alloc(FEATD * sizeof(float));
    float*  ea    = (float*)alloc(OUTC * sizeof(float));
    float*  ec    = (float*)alloc(OUTC * sizeof(float));
    float*  psum  = (float*)alloc((size_t)OUTC * NSLICE * sizeof(float));
    float*  psq   = (float*)alloc((size_t)OUTC * NSLICE * sizeof(float));

    const int CT = 256;

    // 1) bf16 conversions
    cvt_bf16<<<2048, CT, 0, stream>>>(cls, Xbf, MAT);
    for (int e = 0; e < NEXP; ++e)
        cvt_bf16<<<2048, CT, 0, stream>>>(xs[e], Xbf + (size_t)(e + 1) * MAT, MAT);
    cvt_bf16<<<2048, CT, 0, stream>>>(W_re, WreBf, (size_t)FEATD * 8 * FEATD);
    cvt_bf16<<<1024, CT, 0, stream>>>(Wq, WqBf, (size_t)FEATD * FEATD);
    cvt_bf16<<<1024, CT, 0, stream>>>(Wk, WkBf, (size_t)FEATD * FEATD);
    cvt_bf16<<<512,  CT, 0, stream>>>(We, WeBf, (size_t)NHEAD * NEXP * HDIM * HDIM);

    // 2) gating GEMM + bias + quickgelu -> query (f32)
    {
        const int waves = (B_ROWS / 64) * (FEATD / 64);  // 2048
        gemm_gate<<<waves * 32 / CT, CT, 0, stream>>>(Xbf, WreBf, b_re, query);
    }

    // 3) BN over batch for query, apply, convert to bf16
    colstats_partial<<<(FEATD * NSLICE + CT - 1) / CT, CT, 0, stream>>>(query, FEATD, FEATD, psum, psq);
    colstats_final<<<(FEATD + CT - 1) / CT, CT, 0, stream>>>(psum, psq, FEATD, g_re, bb_re, 0, a1, c1);
    bn_apply_bf16<<<2048, CT, 0, stream>>>(query, a1, c1, qbnBf, MAT);

    // 4) q = qbn @ Wq^T   (overwrites the query buffer — pre-BN query no longer needed)
    {
        dim3 grid((B_ROWS / 64) * (FEATD / 64) * 32 / CT, 1);
        gemm_nt<<<grid, CT, 0, stream>>>(qbnBf, 0, WqBf, query, 0, FEATD, FEATD / 64, FEATD);
    }

    // 5) k_e = x_e @ Wk^T for e=0..6, batched over grid.y, interleaved as (b, e, 1024)
    {
        dim3 grid((B_ROWS / 64) * (FEATD / 64) * 32 / CT, NEXP);
        gemm_nt<<<grid, CT, 0, stream>>>(Xbf + MAT, MAT, WkBf, kmat, FEATD, OUTC, FEATD / 64, FEATD);
    }

    // 6) attention logits + softmax -> gates
    attn_softmax<<<(B_ROWS * NHEAD) / CT, CT, 0, stream>>>(query, kmat, gates);

    // 7) expert GEMMs (+bias, quickgelu) -> d_out (pre-BN)
    {
        const int waves = NHEAD * NEXP * (B_ROWS / 32);  // 28672
        gemm_expert<<<waves * 32 / CT, CT, 0, stream>>>(Xbf + MAT, WeBf, be, (float*)d_out);
    }

    // 8) per-column BN stats over d_out, then BN + gate in place
    colstats_partial<<<(OUTC * NSLICE + CT - 1) / CT, CT, 0, stream>>>((float*)d_out, OUTC, OUTC, psum, psq);
    colstats_final<<<(OUTC + CT - 1) / CT, CT, 0, stream>>>(psum, psq, OUTC, g_e, bb_e, 1, ea, ec);
    finalize<<<8192, CT, 0, stream>>>((float*)d_out, ea, ec, gates, (size_t)B_ROWS * OUTC);
}
